// rollout_prediction_58299886076584
// MI455X (gfx1250) — matrix operand order repaired and checked by reading, not compiled, not run on hardware
//
#include <hip/hip_runtime.h>
#include <hip/hip_bf16.h>

// ---------------------------------------------------------------------------
// GRU rollout for MI455X (gfx1250), wave32 + WMMA bf16.
//
// Dims: OBS=256, ACT=128, C=2048, G=3C=6144, P=128, T=4096.
// Fused recurrent matvec: gates = [h, a_t] @ [Wh; Wi]  (K = 2176 = 68 ktiles)
// bf16 weights packed in WMMA-B-tile-native layout; fp32 accumulation.
// x vector staged in LDS per step; weight tiles streamed from L2.
// Scalar (blockIdx-derived) k-half split => no EXEC-masked WMMA.
// ---------------------------------------------------------------------------

#define OBS_DIM   256
#define ACT_DIM   128
#define C_DIM     2048
#define G_DIM     6144          // 3 * C
#define P_DIM     128
#define T_STEPS   4096
#define KTILES    68            // 64 h-ktiles + 4 a-ktiles (32 K each)
#define JTILES_G  384           // 6144 / 16 gate output tiles
#define JTILES_O  16            // 256 / 16 output-head tiles
#define NWG       97            // block 0: output head; 1..48: k-half0; 49..96: k-half1
#define X_LEN     2176          // C + ACT

typedef __attribute__((ext_vector_type(16))) __bf16 v16bf;
typedef __attribute__((ext_vector_type(8)))  __bf16 v8bf;
typedef __attribute__((ext_vector_type(8)))  float  v8f;

// ---- workspace layout (bytes) ---------------------------------------------
constexpr size_t SZ_WPACK  = (size_t)KTILES * JTILES_G * 512 * 2;   // 26,738,688
constexpr size_t OFF_WPACK = 0;
constexpr size_t OFF_WOPACK= OFF_WPACK + SZ_WPACK;
constexpr size_t SZ_WOPACK = (size_t)64 * JTILES_O * 512 * 2;       // 1,048,576
constexpr size_t OFF_HBUF  = OFF_WOPACK + SZ_WOPACK;                // f32[2048]
constexpr size_t OFF_G1A   = OFF_HBUF + C_DIM * 4;                  // f32[6144]
constexpr size_t OFF_G1B   = OFF_G1A + G_DIM * 4;                   // f32[6144]
constexpr size_t OFF_G2    = OFF_G1B + G_DIM * 4;                   // f32[6144]
constexpr size_t OFF_REW   = OFF_G2  + G_DIM * 4;                   // f32[4096]
constexpr size_t OFF_XBUF  = OFF_REW + T_STEPS * 4;                 // bf16[2176]
constexpr size_t OFF_BAR   = OFF_XBUF + 4608;                       // u32[2]

// ---------------------------------------------------------------------------
__device__ __forceinline__ __bf16 f2bf(float f) {
  union { float f; unsigned u; } a; a.f = f;
  unsigned u = a.u + 0x7FFFu + ((a.u >> 16) & 1u);   // round-to-nearest-even
  unsigned short s = (unsigned short)(u >> 16);
  union { unsigned short s; __bf16 b; } c; c.s = s; return c.b;
}

// Broadcast-A tile from LDS-staged x: every row M = x[k0 .. k0+31], in the
// contiguous K order matching the packed B tiles (lanes 0-15: K = 0..15;
// lanes 16-31: K = 16..31).  Two ds_load_b128 per lane.
__device__ __forceinline__ v16bf load_bcast_a(const __bf16* xs, int k0, int lane) {
  int aoff = (lane & 16) ? 16 : 0;
  union { v16bf v; v8bf h[2]; } au;
  au.h[0] = *(const v8bf*)(xs + k0 + aoff);
  au.h[1] = *(const v8bf*)(xs + k0 + aoff + 8);
  return au.v;
}

// device-wide split barrier on ws (persistent grid)
__device__ __forceinline__ void gridbar(unsigned* bar) {
  __threadfence();
  __syncthreads();
  if (threadIdx.x == 0) {
    unsigned g = __atomic_load_n(bar + 1, __ATOMIC_ACQUIRE);
    unsigned prev = atomicAdd(bar, 1u);
    if (prev == (unsigned)(NWG - 1)) {
      __atomic_store_n(bar, 0u, __ATOMIC_RELAXED);
      __atomic_fetch_add(bar + 1, 1u, __ATOMIC_RELEASE);
    } else {
      while (__atomic_load_n(bar + 1, __ATOMIC_ACQUIRE) == g)
        __builtin_amdgcn_s_sleep(2);
    }
  }
  __syncthreads();
  __threadfence();
}

// ---------------------------------------------------------------------------
// Pack [Wh; Wi] (K = 2176) into B-tile-native bf16:
// tile(kt, jt) holds 32x16 values; lane L owns 16 contiguous halves
// (K = (L&16)+e for e=0..15, N = L&15)  -> one 32B load per lane at use site.
__global__ void pack_w_kernel(const float* __restrict__ Wh,
                              const float* __restrict__ Wi,
                              __bf16* __restrict__ wp) {
  int idx = blockIdx.x * 256 + threadIdx.x;
  if (idx >= KTILES * JTILES_G * 512) return;
  int e    = idx & 15;
  int lane = (idx >> 4) & 31;
  int jt   = (idx >> 9) % JTILES_G;
  int kt   = (idx >> 9) / JTILES_G;
  int k    = kt * 32 + (lane & 16) + e;
  int j    = jt * 16 + (lane & 15);
  float v  = (k < C_DIM) ? Wh[(size_t)k * G_DIM + j]
                         : Wi[(size_t)(k - C_DIM) * G_DIM + j];
  wp[idx] = f2bf(v);
}

__global__ void pack_wo_kernel(const float* __restrict__ Wo,
                               __bf16* __restrict__ wo) {
  int idx = blockIdx.x * 256 + threadIdx.x;
  if (idx >= 64 * JTILES_O * 512) return;
  int e    = idx & 15;
  int lane = (idx >> 4) & 31;
  int jt   = (idx >> 9) & 15;
  int kt   = idx >> 13;
  int k    = kt * 32 + (lane & 16) + e;
  int j    = jt * 16 + (lane & 15);
  wo[idx] = f2bf(Wo[(size_t)k * (2 * P_DIM) + j]);
}

// carry0 = obs @ We + be ; x = [carry0, a_0] ; init barrier
__global__ void init_kernel(const float* __restrict__ obs,
                            const float* __restrict__ We,
                            const float* __restrict__ be,
                            const float* __restrict__ act,
                            char* __restrict__ ws) {
  float*  hbuf = (float*)(ws + OFF_HBUF);
  __bf16* xbuf = (__bf16*)(ws + OFF_XBUF);
  unsigned* bar = (unsigned*)(ws + OFF_BAR);
  int tid = blockIdx.x * 256 + threadIdx.x;
  if (tid < C_DIM) {
    float s = be[tid];
    for (int o = 0; o < OBS_DIM; ++o) s += obs[o] * We[(size_t)o * C_DIM + tid];
    hbuf[tid] = s;
    xbuf[tid] = f2bf(s);
  } else if (tid < C_DIM + ACT_DIM) {
    int i = tid - C_DIM;
    xbuf[C_DIM + i] = f2bf(act[i]);
  } else if (tid == C_DIM + ACT_DIM) {
    bar[0] = 0u; bar[1] = 0u;
  }
}

// ---------------------------------------------------------------------------
// Persistent rollout: 2 grid barriers / step.
//   A-slot: blocks 1..96 (768 waves) -> gates matvec via WMMA bf16
//           (blocks 1..48: h-ktiles 0..33; blocks 49..96: h-ktiles 34..63 +
//            a-ktiles 64..67 in a separate accumulator for the n gate);
//           block 0 -> previous step's output head (h @ Wo) + reward.
//   B-slot: gate nonlinearities, h update, next-action bf16 convert.
__global__ void rollout_kernel(const float* __restrict__ obs,
                               const float* __restrict__ act,
                               const float* __restrict__ bh,
                               const float* __restrict__ bo,
                               float* __restrict__ out,
                               char* __restrict__ ws) {
  const __bf16* wp   = (const __bf16*)(ws + OFF_WPACK);
  const __bf16* wo   = (const __bf16*)(ws + OFF_WOPACK);
  float*  hbuf = (float*)(ws + OFF_HBUF);
  float*  g1a  = (float*)(ws + OFF_G1A);
  float*  g1b  = (float*)(ws + OFF_G1B);
  float*  g2   = (float*)(ws + OFF_G2);
  float*  rew  = (float*)(ws + OFF_REW);
  __bf16* xbuf = (__bf16*)(ws + OFF_XBUF);
  unsigned* bar = (unsigned*)(ws + OFF_BAR);

  __shared__ __attribute__((aligned(16))) __bf16 sh_x[X_LEN];
  __shared__ float sh_out[2 * P_DIM];
  __shared__ float sh_red[2 * P_DIM];

  const int lane = threadIdx.x & 31;
  const int wv   = threadIdx.x >> 5;
  const int gtid = blockIdx.x * 256 + threadIdx.x;

  // scalar (uniform) role selection: compiler sees blockIdx in SGPRs
  const int  blk    = (int)blockIdx.x;
  const bool isHead = (blk == 0);
  const int  b      = blk - 1;             // 0..95 for gate blocks
  const bool half1  = (b >= 48);           // scalar branch, no EXEC masking
  const int  jb     = half1 ? (b - 48) : b;

  for (int t = 0; t < T_STEPS; ++t) {
    // stage x = [h(t), a(t)] into LDS (written by previous phase B / init)
    for (int i = threadIdx.x; i < X_LEN / 8; i += 256)
      ((v8bf*)sh_x)[i] = ((const v8bf*)xbuf)[i];
    __syncthreads();

    // ---------------- phase A slot ----------------
    if (!isHead) {
      int jt = jb * 8 + wv;                // 0..383
      v8f accH = {0.f,0.f,0.f,0.f,0.f,0.f,0.f,0.f};
      if (!half1) {
        // h-ktiles 0..33, constant trip count, single chained accumulator
        for (int kt = 0; kt < 34; ++kt) {
          const __bf16* bt = wp + (((size_t)kt * JTILES_G + jt) << 9) + (lane << 4);
          __builtin_prefetch(wp + (((size_t)(kt + 1) * JTILES_G + jt) << 9) + (lane << 4), 0, 1);
          v16bf bm = *(const v16bf*)bt;
          v16bf am = load_bcast_a(sh_x, kt << 5, lane);
          accH = __builtin_amdgcn_wmma_f32_16x16x32_bf16(false, am, false, bm,
                                                         (short)0, accH, false, false);
        }
        if (lane < 16) g1a[jt * 16 + lane] = accH[0];
      } else {
        // h-ktiles 34..63
        for (int kt = 34; kt < 64; ++kt) {
          const __bf16* bt = wp + (((size_t)kt * JTILES_G + jt) << 9) + (lane << 4);
          __builtin_prefetch(wp + (((size_t)(kt + 1) * JTILES_G + jt) << 9) + (lane << 4), 0, 1);
          v16bf bm = *(const v16bf*)bt;
          v16bf am = load_bcast_a(sh_x, kt << 5, lane);
          accH = __builtin_amdgcn_wmma_f32_16x16x32_bf16(false, am, false, bm,
                                                         (short)0, accH, false, false);
        }
        // a-ktiles 64..67: separate accumulator (i_n needed apart from h_n)
        v8f accA = {0.f,0.f,0.f,0.f,0.f,0.f,0.f,0.f};
        for (int kt = 64; kt < 68; ++kt) {
          const __bf16* bt = wp + (((size_t)kt * JTILES_G + jt) << 9) + (lane << 4);
          v16bf bm = *(const v16bf*)bt;
          v16bf am = load_bcast_a(sh_x, kt << 5, lane);
          accA = __builtin_amdgcn_wmma_f32_16x16x32_bf16(false, am, false, bm,
                                                         (short)0, accA, false, false);
        }
        if (lane < 16) {
          int j = jt * 16 + lane;
          g1b[j] = accH[0];
          g2[j]  = accA[0];
        }
      }
    } else if (t > 0) {
      // output head for step (t-1): mu/logvar = h(t) @ Wo + bo, plus reward
      int tprev = t - 1;
      for (int rep = 0; rep < 2; ++rep) {
        int jt = wv * 2 + rep;                   // 0..15
        v8f acc = {0.f,0.f,0.f,0.f,0.f,0.f,0.f,0.f};
        for (int kt = 0; kt < 64; ++kt) {
          v16bf bm = *(const v16bf*)(wo + (((size_t)kt * JTILES_O + jt) << 9) + (lane << 4));
          v16bf am = load_bcast_a(sh_x, kt << 5, lane);
          acc = __builtin_amdgcn_wmma_f32_16x16x32_bf16(false, am, false, bm,
                                                        (short)0, acc, false, false);
        }
        if (lane < 16) sh_out[jt * 16 + lane] = acc[0];
      }
      __syncthreads();
      int j = threadIdx.x;                        // 0..255
      float val = sh_out[j] + bo[j];
      float term;
      if (j < P_DIM) {
        out[(size_t)tprev * P_DIM + j] = val;                 // mu
        float ma = val + obs[OBS_DIM - P_DIM + j];            // mu_abs
        float ap = act[(size_t)tprev * ACT_DIM + j];
        term = ma * ma + 1e-3f * ap * ap;
      } else {
        out[(size_t)T_STEPS * P_DIM + (size_t)tprev * P_DIM + (j - P_DIM)] = val; // log_var
        term = __expf(val);
      }
      sh_red[j] = term;
      __syncthreads();
      for (int s = 128; s > 0; s >>= 1) {
        if (threadIdx.x < s) sh_red[threadIdx.x] += sh_red[threadIdx.x + s];
        __syncthreads();
      }
      if (threadIdx.x == 0) rew[tprev] = -sh_red[0];
    }

    gridbar(bar);

    // ---------------- phase B: gates + h update ----------------
    if (gtid < C_DIM) {
      int c = gtid;
      float s_r = g1a[c]            + g1b[c]            + g2[c]            + bh[c];
      float s_z = g1a[c + C_DIM]    + g1b[c + C_DIM]    + g2[c + C_DIM]    + bh[c + C_DIM];
      float hn  = g1a[c + 2*C_DIM]  + g1b[c + 2*C_DIM]                     + bh[c + 2*C_DIM];
      float in_ = g2[c + 2*C_DIM];
      float r = 1.0f / (1.0f + __expf(-s_r));
      float z = 1.0f / (1.0f + __expf(-s_z));
      float n = tanhf(in_ + r * hn);
      float h_new = (1.0f - z) * n + z * hbuf[c];
      hbuf[c] = h_new;
      xbuf[c] = f2bf(h_new);
    } else if (gtid < C_DIM + ACT_DIM && t + 1 < T_STEPS) {
      int i = gtid - C_DIM;
      xbuf[C_DIM + i] = f2bf(act[(size_t)(t + 1) * ACT_DIM + i]);
    }

    gridbar(bar);
  }

  // ---------------- final output head: step T-1 ----------------
  if (isHead) {
    // re-stage x (h(T) was written by the last phase B)
    for (int i = threadIdx.x; i < X_LEN / 8; i += 256)
      ((v8bf*)sh_x)[i] = ((const v8bf*)xbuf)[i];
    __syncthreads();
    int tprev = T_STEPS - 1;
    for (int rep = 0; rep < 2; ++rep) {
      int jt = wv * 2 + rep;
      v8f acc = {0.f,0.f,0.f,0.f,0.f,0.f,0.f,0.f};
      for (int kt = 0; kt < 64; ++kt) {
        v16bf bm = *(const v16bf*)(wo + (((size_t)kt * JTILES_O + jt) << 9) + (lane << 4));
        v16bf am = load_bcast_a(sh_x, kt << 5, lane);
        acc = __builtin_amdgcn_wmma_f32_16x16x32_bf16(false, am, false, bm,
                                                      (short)0, acc, false, false);
      }
      if (lane < 16) sh_out[jt * 16 + lane] = acc[0];
    }
    __syncthreads();
    int j = threadIdx.x;
    float val = sh_out[j] + bo[j];
    float term;
    if (j < P_DIM) {
      out[(size_t)tprev * P_DIM + j] = val;
      float ma = val + obs[OBS_DIM - P_DIM + j];
      float ap = act[(size_t)tprev * ACT_DIM + j];
      term = ma * ma + 1e-3f * ap * ap;
    } else {
      out[(size_t)T_STEPS * P_DIM + (size_t)tprev * P_DIM + (j - P_DIM)] = val;
      term = __expf(val);
    }
    sh_red[j] = term;
    __syncthreads();
    for (int s = 128; s > 0; s >>= 1) {
      if (threadIdx.x < s) sh_red[threadIdx.x] += sh_red[threadIdx.x + s];
      __syncthreads();
    }
    if (threadIdx.x == 0) rew[tprev] = -sh_red[0];
  }
}

__global__ void reduce_kernel(const char* __restrict__ ws, float* __restrict__ out) {
  const float* rew = (const float*)(ws + OFF_REW);
  __shared__ float sh[256];
  float s = 0.0f;
  for (int i = threadIdx.x; i < T_STEPS; i += 256) s += rew[i];
  sh[threadIdx.x] = s;
  __syncthreads();
  for (int k = 128; k > 0; k >>= 1) {
    if (threadIdx.x < k) sh[threadIdx.x] += sh[threadIdx.x + k];
    __syncthreads();
  }
  if (threadIdx.x == 0) out[(size_t)2 * T_STEPS * P_DIM] = sh[0];
}

// ---------------------------------------------------------------------------
extern "C" void kernel_launch(void* const* d_in, const int* in_sizes, int n_in,
                              void* d_out, int out_size, void* d_ws, size_t ws_size,
                              hipStream_t stream) {
  const float* obs = (const float*)d_in[0];
  const float* act = (const float*)d_in[1];
  const float* We  = (const float*)d_in[2];
  const float* be  = (const float*)d_in[3];
  const float* Wi  = (const float*)d_in[4];
  const float* Wh  = (const float*)d_in[5];
  const float* bh  = (const float*)d_in[6];
  const float* Wo  = (const float*)d_in[7];
  const float* bo  = (const float*)d_in[8];
  float* out = (float*)d_out;
  char*  ws  = (char*)d_ws;

  __bf16* wp = (__bf16*)(ws + OFF_WPACK);
  __bf16* wq = (__bf16*)(ws + OFF_WOPACK);

  pack_w_kernel <<<(KTILES * JTILES_G * 512) / 256, 256, 0, stream>>>(Wh, Wi, wp);
  pack_wo_kernel<<<(64 * JTILES_O * 512) / 256, 256, 0, stream>>>(Wo, wq);
  init_kernel   <<<9, 256, 0, stream>>>(obs, We, be, act, ws);
  rollout_kernel<<<NWG, 256, 0, stream>>>(obs, act, bh, bo, out, ws);
  reduce_kernel <<<1, 256, 0, stream>>>(ws, out);
}
